// ARTEMIS_36026185678959
// MI455X (gfx1250) — matrix operand order repaired and checked by reading, not compiled, not run on hardware
//
#include <hip/hip_runtime.h>
#include <hip/hip_bf16.h>
#include <math.h>

#define NNODES  50000
#define NEDGES  800000
#define ETOT    850000   // edges + self loops
#define NGRAPHS 64
#define INCH    32
#define HIDN    128
#define NHEADS  4
#define DHD     32
#define LRELU_S 0.2f
#define ORD_NEG_INF ((int)0x807FFFFF)   // f2ord(-inf)

typedef __attribute__((ext_vector_type(16))) _Float16 v16h;
typedef __attribute__((ext_vector_type(8)))  _Float16 v8h;
typedef __attribute__((ext_vector_type(8)))  float    v8f;

static inline int cdiv(long long a, long long b){ return (int)((a + b - 1) / b); }

__device__ __forceinline__ float gelu_f(float x){ return 0.5f * x * (1.0f + erff(x * 0.70710678118654752f)); }
__device__ __forceinline__ int   f2ord(float f){ int i = __float_as_int(f); return (i >= 0) ? i : (i ^ 0x7FFFFFFF); }
__device__ __forceinline__ float ord2f(int i){ return __int_as_float((i >= 0) ? i : (i ^ 0x7FFFFFFF)); }
__device__ __forceinline__ void  edge_sd(const int* __restrict__ ei, int e, int& s, int& d){
  if (e < NEDGES){ s = ei[e]; d = ei[NEDGES + e]; } else { s = e - NEDGES; d = s; }
}

// ---------------------------------------------------------------- fills
__global__ void k_fill_f32(float* p, float v, int n){
  int i = blockIdx.x * blockDim.x + threadIdx.x; if (i < n) p[i] = v;
}
__global__ void k_fill_i32(int* p, int v, int n){
  int i = blockIdx.x * blockDim.x + threadIdx.x; if (i < n) p[i] = v;
}

// ---------------------------------------------------------------- WMMA GEMM
// C[nrows, M] = epi( (A + alpha*A2)[nrows, K] @ B[K, M] + bias )
// EPI: 0 = none, 1 = +bias, 2 = +bias->gelu, 3 = +bias->tanh
// K, M compile-time so accumulators stay in VGPRs (no scratch spill) and all
// LDS fragment addresses fold to base + immediate offsets.
// Block = 256 threads (8 waves). Wave w covers rows [blk*128 + w*16, +16).
// B staged in LDS as f16, layout [M][K] so each lane's 16-element fragment is
// two 16-byte runs (K offsets +0 and +16) -> ds_load_b128 x2.
template<int EPI, int K, int M>
__global__ void k_wmma_gemm(const float* __restrict__ A, const float* __restrict__ A2, float alpha,
                            const float* __restrict__ B, const float* __restrict__ bias,
                            float* __restrict__ C, int nrows)
{
  constexpr int KT = K / 32;   // K-tiles of 32
  constexpr int NT = M / 16;   // N-tiles of 16

  extern __shared__ char smem_raw[];
  _Float16* sB   = (_Float16*)smem_raw;
  const v8h* sBv = (const v8h*)smem_raw;

  const int tid = threadIdx.x;
  for (int i = tid; i < M * K; i += 256) {
    int n = i / K; int kk = i - n * K;
    sB[i] = (_Float16)B[(size_t)kk * M + n];
  }
  __syncthreads();

  const int wave = tid >> 5;
  const int lane = tid & 31;
  const int row0 = blockIdx.x * 128 + wave * 16;
  if (row0 >= nrows) return;            // wave-uniform: EXEC stays all-ones inside

  const int khalf = lane >> 4;          // 0 or 1
  const int l15   = lane & 15;
  int m = row0 + l15; if (m > nrows - 1) m = nrows - 1;   // clamp ragged tail

  v8f acc[NT];
  const v8f vzero = {0.f,0.f,0.f,0.f,0.f,0.f,0.f,0.f};
#pragma unroll
  for (int i = 0; i < NT; ++i) acc[i] = vzero;

  const bool hasA2 = (A2 != nullptr);
  const float* ar  = A + (size_t)m * K;
  const float* ar2 = hasA2 ? (A2 + (size_t)m * K) : ar;
  const int b1 = khalf * 8, b2 = 16 + khalf * 8;

#pragma unroll
  for (int kt = 0; kt < KT; ++kt) {
    // A fragment (ISA 7.12.2 16-bit A 16x32): lane holds M=l15;
    // elems 0..7  -> K = kt*32 + khalf*8 + j
    // elems 8..15 -> K = kt*32 + 16 + khalf*8 + j
    v16h a;
    const float* ap  = ar  + kt * 32;
    const float* ap2 = ar2 + kt * 32;
#pragma unroll
    for (int j = 0; j < 8; ++j) {
      float v0 = ap[b1 + j]; if (hasA2) v0 += alpha * ap2[b1 + j];
      float v1 = ap[b2 + j]; if (hasA2) v1 += alpha * ap2[b2 + j];
      a[j]     = (_Float16)v0;
      a[8 + j] = (_Float16)v1;
    }
#pragma unroll
    for (int nt = 0; nt < NT; ++nt) {
      // B fragment: lane holds N = nt*16 + l15; elems 0..7 K = kt*32 + khalf*8 + j, elems 8..15 K = kt*32 + 16 + khalf*8 + j
      const int off8 = (((nt * 16 + l15) * K) + kt * 32 + khalf * 8) >> 3;
      v8h lo = sBv[off8], hi = sBv[off8 + 2];
      v16h b;
#pragma unroll
      for (int j = 0; j < 8; ++j) { b[j] = lo[j]; b[8 + j] = hi[j]; }
      acc[nt] = __builtin_amdgcn_wmma_f32_16x16x32_f16(
          false, a, false, b, (short)0, acc[nt], false, false);
    }
  }

  // D layout: lanes 0-15: VGPR r -> M=r, N=lane; lanes 16-31: M=8+r, N=lane-16
#pragma unroll
  for (int nt = 0; nt < NT; ++nt) {
    const int col = nt * 16 + l15;
#pragma unroll
    for (int r = 0; r < 8; ++r) {
      int row = row0 + khalf * 8 + r;
      if (row < nrows) {
        float v = acc[nt][r];
        if (EPI >= 1) v += bias[col];
        if (EPI == 2) v = gelu_f(v);
        if (EPI == 3) v = tanhf(v);
        C[(size_t)row * M + col] = v;
      }
    }
  }
}

// ---------------------------------------------------------------- GAT pieces
__global__ void k_esed(const float* __restrict__ xh, const float* __restrict__ as_,
                       const float* __restrict__ ad_, float* __restrict__ es,
                       float* __restrict__ ed, int n)
{
  int t = blockIdx.x * blockDim.x + threadIdx.x;
  if (t >= n * NHEADS) return;
  int node = t >> 2, hh = t & 3;
  const float* xr = xh + (size_t)node * HIDN + hh * DHD;
  float s0 = 0.f, s1 = 0.f;
  for (int d = 0; d < DHD; ++d) { float v = xr[d]; s0 += v * as_[hh*DHD+d]; s1 += v * ad_[hh*DHD+d]; }
  es[t] = s0; ed[t] = s1;
}

__global__ void k_edge_max(const int* __restrict__ ei, const float* __restrict__ es,
                           const float* __restrict__ ed, int* __restrict__ menc)
{
  long long t = (long long)blockIdx.x * blockDim.x + threadIdx.x;
  if (t >= (long long)ETOT * NHEADS) return;
  int e = (int)(t >> 2), hh = (int)(t & 3), s, d;
  edge_sd(ei, e, s, d);
  float v = es[s * NHEADS + hh] + ed[d * NHEADS + hh];
  v = (v > 0.f) ? v : LRELU_S * v;
  atomicMax(&menc[d * NHEADS + hh], f2ord(v));
}

__global__ void k_edge_w(const int* __restrict__ ei, const float* __restrict__ es,
                         const float* __restrict__ ed, const int* __restrict__ menc,
                         float* __restrict__ den, float* __restrict__ wedge)
{
  long long t = (long long)blockIdx.x * blockDim.x + threadIdx.x;
  if (t >= (long long)ETOT * NHEADS) return;
  int e = (int)(t >> 2), hh = (int)(t & 3), s, d;
  edge_sd(ei, e, s, d);
  float v = es[s * NHEADS + hh] + ed[d * NHEADS + hh];
  v = (v > 0.f) ? v : LRELU_S * v;
  float mv = ord2f(menc[d * NHEADS + hh]);
  if (!isfinite(mv)) mv = 0.f;
  float w = expf(v - mv);
  wedge[t] = w;
  atomicAdd(&den[d * NHEADS + hh], w);
}

__global__ void k_edge_scatter(const int* __restrict__ ei, const float* __restrict__ xh,
                               const float* __restrict__ wedge, const float* __restrict__ den,
                               float* __restrict__ out)
{
  long long t = (long long)blockIdx.x * blockDim.x + threadIdx.x;
  if (t >= (long long)ETOT * HIDN) return;
  int e = (int)(t >> 7), f = (int)(t & 127), hh = f >> 5, s, d;
  edge_sd(ei, e, s, d);
  float c = wedge[(size_t)e * NHEADS + hh] / (den[d * NHEADS + hh] + 1e-16f);
  atomicAdd(&out[(size_t)d * HIDN + f], c * xh[(size_t)s * HIDN + f]);
}

__global__ void k_gelu_ln(float* __restrict__ h, const float* __restrict__ g,
                          const float* __restrict__ b, int n)
{
  int node = blockIdx.x * 8 + (threadIdx.x >> 5);
  int lane = threadIdx.x & 31;
  if (node >= n) return;
  float v[4]; float s = 0.f, s2 = 0.f;
#pragma unroll
  for (int j = 0; j < 4; ++j) {
    float x = h[(size_t)node * HIDN + lane + 32 * j];
    x = gelu_f(x);
    v[j] = x; s += x; s2 += x * x;
  }
  for (int o = 16; o >= 1; o >>= 1) { s += __shfl_xor(s, o, 32); s2 += __shfl_xor(s2, o, 32); }
  float mu = s * (1.f / HIDN);
  float var = s2 * (1.f / HIDN) - mu * mu;
  float r = rsqrtf(var + 1e-5f);
#pragma unroll
  for (int j = 0; j < 4; ++j) {
    int f = lane + 32 * j;
    h[(size_t)node * HIDN + f] = (v[j] - mu) * r * g[f] + b[f];
  }
}

// ---------------------------------------------------------------- ODE helper
__global__ void k_axpby(float* __restrict__ dst, const float* __restrict__ x, float a,
                        const float* __restrict__ y, float b, int n)
{
  int i = blockIdx.x * blockDim.x + threadIdx.x;
  if (i < n) dst[i] = a * x[i] + b * y[i];
}

// ---------------------------------------------------------------- hops
__global__ void k_deg(const int* __restrict__ ei, float* __restrict__ deg)
{
  int e = blockIdx.x * blockDim.x + threadIdx.x;
  if (e >= ETOT) return;
  int s, d; edge_sd(ei, e, s, d);
  atomicAdd(&deg[d], 1.0f);
}
__global__ void k_hop_scatter(const int* __restrict__ ei, const float* __restrict__ cur,
                              float* __restrict__ nxt)
{
  long long t = (long long)blockIdx.x * blockDim.x + threadIdx.x;
  if (t >= (long long)ETOT * HIDN) return;
  int e = (int)(t >> 7), f = (int)(t & 127), s, d;
  edge_sd(ei, e, s, d);
  atomicAdd(&nxt[(size_t)d * HIDN + f], cur[(size_t)s * HIDN + f]);
}
__global__ void k_hop_div(float* __restrict__ p, const float* __restrict__ deg, int n)
{
  int t = blockIdx.x * blockDim.x + threadIdx.x;
  if (t >= n * HIDN) return;
  p[t] /= fmaxf(deg[t >> 7], 1.0f);
}
__global__ void k_dot_qb(const float* __restrict__ tmp, const float* __restrict__ qb,
                         float* __restrict__ s, int n)
{
  int node = blockIdx.x * blockDim.x + threadIdx.x;
  if (node >= n) return;
  float acc = 0.f;
  const float* r = tmp + (size_t)node * HIDN;
  for (int f = 0; f < HIDN; ++f) acc += r[f] * qb[f];
  s[node] = acc;
}
__global__ void k_hop_combine(float* __restrict__ h0, const float* __restrict__ h1,
                              const float* __restrict__ h2, const float* __restrict__ h3,
                              const float* __restrict__ s, int n)
{
  long long t = (long long)blockIdx.x * blockDim.x + threadIdx.x;
  if (t >= (long long)n * HIDN) return;
  int node = (int)(t >> 7);
  float s0 = s[node], s1 = s[n + node], s2 = s[2 * n + node], s3 = s[3 * n + node];
  float mx = fmaxf(fmaxf(s0, s1), fmaxf(s2, s3));
  float e0 = expf(s0 - mx), e1 = expf(s1 - mx), e2 = expf(s2 - mx), e3 = expf(s3 - mx);
  float inv = 1.f / (e0 + e1 + e2 + e3);
  h0[t] = (e0 * h0[t] + e1 * h1[t] + e2 * h2[t] + e3 * h3[t]) * inv;
}

// ---------------------------------------------------------------- pooling / readout
__global__ void k_pool_cnt(const int* __restrict__ batch, float* __restrict__ cnt, int n)
{
  int i = blockIdx.x * blockDim.x + threadIdx.x;
  if (i < n) atomicAdd(&cnt[batch[i]], 1.0f);
}
__global__ void k_pool_accum(const float* __restrict__ h, const int* __restrict__ batch,
                             float* __restrict__ mean, int* __restrict__ mxe, int n)
{
  long long t = (long long)blockIdx.x * blockDim.x + threadIdx.x;
  if (t >= (long long)n * HIDN) return;
  int node = (int)(t >> 7), f = (int)(t & 127);
  int g = batch[node];
  float v = h[t];
  atomicAdd(&mean[g * HIDN + f], v);
  atomicMax(&mxe[g * HIDN + f], f2ord(v));
}
__global__ void k_pool_final(const float* __restrict__ mean, const int* __restrict__ mxe,
                             const float* __restrict__ cnt, float* __restrict__ hg)
{
  int t = blockIdx.x * blockDim.x + threadIdx.x;
  if (t >= NGRAPHS * HIDN) return;
  int g = t >> 7, f = t & 127;
  hg[g * 2 * HIDN + f] = mean[t] / fmaxf(cnt[g], 1.0f);
  float mx = ord2f(mxe[t]);
  hg[g * 2 * HIDN + HIDN + f] = isfinite(mx) ? mx : 0.0f;
}
__global__ void k_classifier(const float* __restrict__ r2, const float* __restrict__ Wc,
                             const float* __restrict__ bc, float* __restrict__ out)
{
  int t = blockIdx.x * blockDim.x + threadIdx.x;
  if (t >= NGRAPHS * 2) return;
  int g = t >> 1, c = t & 1;
  float acc = bc[c];
  for (int j = 0; j < 64; ++j) acc += r2[g * 64 + j] * Wc[j * 2 + c];
  out[g * 2 + c] = acc;
}

// ---------------------------------------------------------------- host
template<int EPI, int K, int M>
static inline void launch_gemm(const float* A, const float* A2, float alpha,
                               const float* B, const float* bias, float* C,
                               int nrows, hipStream_t stream)
{
  dim3 grid(cdiv(nrows, 128));
  size_t lds = (size_t)M * K * sizeof(_Float16);
  k_wmma_gemm<EPI, K, M><<<grid, 256, lds, stream>>>(A, A2, alpha, B, bias, C, nrows);
}

extern "C" void kernel_launch(void* const* d_in, const int* in_sizes, int n_in,
                              void* d_out, int out_size, void* d_ws, size_t ws_size,
                              hipStream_t stream)
{
  (void)in_sizes; (void)n_in; (void)out_size; (void)ws_size;
  const float* x     = (const float*)d_in[0];
  const int*   ei    = (const int*)  d_in[1];
  /* d_in[2] timestamps: unused by the reference */
  const int*   batch = (const int*)  d_in[3];
  const float* Wp  = (const float*)d_in[4];  const float* bp  = (const float*)d_in[5];
  const float* Wg1 = (const float*)d_in[6];  const float* a1s = (const float*)d_in[7];
  const float* a1d = (const float*)d_in[8];  const float* g1  = (const float*)d_in[9];
  const float* be1 = (const float*)d_in[10];
  const float* Wg2 = (const float*)d_in[11]; const float* a2s = (const float*)d_in[12];
  const float* a2d = (const float*)d_in[13]; const float* g2  = (const float*)d_in[14];
  const float* be2 = (const float*)d_in[15];
  const float* Wo1 = (const float*)d_in[16]; const float* bo1 = (const float*)d_in[17];
  const float* Wo2 = (const float*)d_in[18]; const float* bo2 = (const float*)d_in[19];
  const float* Wb  = (const float*)d_in[20]; const float* qb  = (const float*)d_in[21];
  const float* Wr1 = (const float*)d_in[22]; const float* br1 = (const float*)d_in[23];
  const float* Wr2 = (const float*)d_in[24]; const float* br2 = (const float*)d_in[25];
  const float* Wc  = (const float*)d_in[26]; const float* bc  = (const float*)d_in[27];

  // workspace carve-up (floats)
  const size_t NH = (size_t)NNODES * HIDN;     // 6.4M
  float* ws   = (float*)d_ws;
  float* t    = ws;             // scratch GEMM out / hop3
  float* kbuf = ws + NH;        // ODE k / hop1
  float* acc  = ws + 2 * NH;    // ODE accumulator / hop2
  float* h    = ws + 3 * NH;    // node state
  float* tmp2 = ws + 4 * NH;    // hop-attention GEMM out
  float* es   = ws + 5 * NH;                        // N*4
  float* ed   = es + (size_t)NNODES * NHEADS;       // N*4
  int*   menc = (int*)(ed + (size_t)NNODES * NHEADS);
  float* den  = (float*)menc + (size_t)NNODES * NHEADS;
  float* wedge= den + (size_t)NNODES * NHEADS;      // ETOT*4
  float* sbuf = wedge + (size_t)ETOT * NHEADS;      // 4*N
  float* deg  = sbuf + 4 * (size_t)NNODES;          // N
  float* cnt  = deg + NNODES;                       // 64
  float* meanb= cnt + NGRAPHS;                      // 64*128
  int*   mxe  = (int*)(meanb + NGRAPHS * HIDN);     // 64*128
  float* hg   = (float*)mxe + NGRAPHS * HIDN;       // 64*256
  float* r1   = hg + NGRAPHS * 2 * HIDN;            // 64*128
  float* r2   = r1 + NGRAPHS * HIDN;                // 64*64
  float* zb   = r2 + NGRAPHS * 64;                  // 256 zero bias

  const int TB = 256;
  k_fill_f32<<<1, TB, 0, stream>>>(zb, 0.f, 256);

  // ---- encoder projection: h = gelu(x @ Wp + bp)
  launch_gemm<2, INCH, HIDN>(x, nullptr, 0.f, Wp, bp, h, NNODES, stream);

  // ---- two GAT layers
  const float* Wg[2]  = {Wg1, Wg2};
  const float* asp[2] = {a1s, a2s};
  const float* adp[2] = {a1d, a2d};
  const float* gp[2]  = {g1, g2};
  const float* bep[2] = {be1, be2};
  for (int L = 0; L < 2; ++L) {
    launch_gemm<0, HIDN, HIDN>(h, nullptr, 0.f, Wg[L], zb, t, NNODES, stream);   // xh
    k_esed<<<cdiv((long long)NNODES * NHEADS, TB), TB, 0, stream>>>(t, asp[L], adp[L], es, ed, NNODES);
    k_fill_i32<<<cdiv((long long)NNODES * NHEADS, TB), TB, 0, stream>>>(menc, ORD_NEG_INF, NNODES * NHEADS);
    k_fill_f32<<<cdiv((long long)NNODES * NHEADS, TB), TB, 0, stream>>>(den, 0.f, NNODES * NHEADS);
    k_edge_max<<<cdiv((long long)ETOT * NHEADS, TB), TB, 0, stream>>>(ei, es, ed, menc);
    k_edge_w<<<cdiv((long long)ETOT * NHEADS, TB), TB, 0, stream>>>(ei, es, ed, menc, den, wedge);
    k_fill_f32<<<cdiv((long long)NH, TB), TB, 0, stream>>>(h, 0.f, (int)NH);
    k_edge_scatter<<<cdiv((long long)ETOT * HIDN, TB), TB, 0, stream>>>(ei, t, wedge, den, h);
    k_gelu_ln<<<cdiv(NNODES, 8), TB, 0, stream>>>(h, gp[L], bep[L], NNODES);
  }

  // ---- neural ODE, RK4 x4; f(z) = tanh(z@Wo1+bo1)@Wo2 + bo2
  const float dt = 0.25f;
  for (int step = 0; step < 4; ++step) {
    // k1
    launch_gemm<3, HIDN, HIDN>(h, nullptr, 0.f, Wo1, bo1, t, NNODES, stream);
    launch_gemm<1, HIDN, HIDN>(t, nullptr, 0.f, Wo2, bo2, kbuf, NNODES, stream);
    k_axpby<<<cdiv((long long)NH, TB), TB, 0, stream>>>(acc, h, 1.f, kbuf, dt / 6.f, (int)NH);
    // k2  (z = h + 0.5*dt*k1 fused into A-load)
    launch_gemm<3, HIDN, HIDN>(h, kbuf, 0.5f * dt, Wo1, bo1, t, NNODES, stream);
    launch_gemm<1, HIDN, HIDN>(t, nullptr, 0.f, Wo2, bo2, kbuf, NNODES, stream);
    k_axpby<<<cdiv((long long)NH, TB), TB, 0, stream>>>(acc, acc, 1.f, kbuf, dt / 3.f, (int)NH);
    // k3
    launch_gemm<3, HIDN, HIDN>(h, kbuf, 0.5f * dt, Wo1, bo1, t, NNODES, stream);
    launch_gemm<1, HIDN, HIDN>(t, nullptr, 0.f, Wo2, bo2, kbuf, NNODES, stream);
    k_axpby<<<cdiv((long long)NH, TB), TB, 0, stream>>>(acc, acc, 1.f, kbuf, dt / 3.f, (int)NH);
    // k4
    launch_gemm<3, HIDN, HIDN>(h, kbuf, dt, Wo1, bo1, t, NNODES, stream);
    launch_gemm<1, HIDN, HIDN>(t, nullptr, 0.f, Wo2, bo2, kbuf, NNODES, stream);
    k_axpby<<<cdiv((long long)NH, TB), TB, 0, stream>>>(acc, acc, 1.f, kbuf, dt / 6.f, (int)NH);
    // h = acc
    k_axpby<<<cdiv((long long)NH, TB), TB, 0, stream>>>(h, acc, 1.f, acc, 0.f, (int)NH);
  }

  // ---- multi-hop mean aggregation + hop attention
  k_fill_f32<<<cdiv(NNODES, TB), TB, 0, stream>>>(deg, 0.f, NNODES);
  k_deg<<<cdiv(ETOT, TB), TB, 0, stream>>>(ei, deg);
  float* hops[4] = { h, kbuf, acc, t };
  for (int j = 1; j <= 3; ++j) {
    k_fill_f32<<<cdiv((long long)NH, TB), TB, 0, stream>>>(hops[j], 0.f, (int)NH);
    k_hop_scatter<<<cdiv((long long)ETOT * HIDN, TB), TB, 0, stream>>>(ei, hops[j - 1], hops[j]);
    k_hop_div<<<cdiv((long long)NH, TB), TB, 0, stream>>>(hops[j], deg, NNODES);
  }
  for (int j = 0; j < 4; ++j) {
    launch_gemm<3, HIDN, HIDN>(hops[j], nullptr, 0.f, Wb, zb, tmp2, NNODES, stream);  // tanh(Hs@Wb)
    k_dot_qb<<<cdiv(NNODES, TB), TB, 0, stream>>>(tmp2, qb, sbuf + (size_t)j * NNODES, NNODES);
  }
  k_hop_combine<<<cdiv((long long)NH, TB), TB, 0, stream>>>(h, kbuf, acc, t, sbuf, NNODES);

  // ---- per-graph pooling
  k_fill_f32<<<1, TB, 0, stream>>>(cnt, 0.f, NGRAPHS);
  k_fill_f32<<<cdiv(NGRAPHS * HIDN, TB), TB, 0, stream>>>(meanb, 0.f, NGRAPHS * HIDN);
  k_fill_i32<<<cdiv(NGRAPHS * HIDN, TB), TB, 0, stream>>>(mxe, ORD_NEG_INF, NGRAPHS * HIDN);
  k_pool_cnt<<<cdiv(NNODES, TB), TB, 0, stream>>>(batch, cnt, NNODES);
  k_pool_accum<<<cdiv((long long)NH, TB), TB, 0, stream>>>(h, batch, meanb, mxe, NNODES);
  k_pool_final<<<cdiv(NGRAPHS * HIDN, TB), TB, 0, stream>>>(meanb, mxe, cnt, hg);

  // ---- readout MLP + classifier
  launch_gemm<2, 2 * HIDN, HIDN>(hg, nullptr, 0.f, Wr1, br1, r1, NGRAPHS, stream);
  launch_gemm<2, HIDN, 64>(r1, nullptr, 0.f, Wr2, br2, r2, NGRAPHS, stream);
  k_classifier<<<1, 128, 0, stream>>>(r2, Wc, bc, (float*)d_out);
}
